// HardCausalDeepONet_9818295239411
// MI455X (gfx1250) — matrix operand order repaired and checked by reading, not compiled, not run on hardware
//
#include <hip/hip_runtime.h>
#include <hip/hip_bf16.h>

typedef _Float16 h16;
typedef __attribute__((ext_vector_type(16))) _Float16 v16h;
typedef __attribute__((ext_vector_type(8)))  float    v8f;

#define PI_F 3.14159265358979323846f

// ---------------- prep kernels ----------------

__global__ void k_f32_to_f16(const float* __restrict__ src, h16* __restrict__ dst, int n) {
    int i = blockIdx.x * blockDim.x + threadIdx.x;
    if (i < n) dst[i] = (h16)src[i];
}

// convert rows x srcCols f32 -> rows x dstCols f16, zero-padding extra columns
__global__ void k_pad_convert(const float* __restrict__ src, h16* __restrict__ dst,
                              int rows, int srcCols, int dstCols) {
    int i = blockIdx.x * blockDim.x + threadIdx.x;
    int total = rows * dstCols;
    if (i >= total) return;
    int r = i / dstCols, c = i % dstCols;
    dst[i] = (c < srcCols) ? (h16)src[r * srcCols + c] : (h16)0.f;
}

// Wf: (2*NM) x S row-major; row k<NM: cos(pi*(k+1)*xs), k>=NM: sin(pi*(k-NM+1)*xs)
__global__ void k_build_trig(h16* __restrict__ Wf, int S, int NM) {
    int i = blockIdx.x * blockDim.x + threadIdx.x;
    int total = 2 * NM * S;
    if (i >= total) return;
    int k = i / S, s = i % S;
    float xs   = -1.f + 2.f * (float)s / (float)(S - 1);
    float mode = (float)((k < NM ? k : k - NM) + 1);
    float ang  = PI_F * mode * xs;
    float v    = (k < NM) ? __cosf(ang) : __sinf(ang);
    Wf[i] = (h16)v;
}

// trunk feature rows: [x, t, sqrt(max(t^2-x^2,0)), causal, 0...0] (KP cols), plus causal f32
__global__ void k_build_tfeat(const float* __restrict__ x, const float* __restrict__ t,
                              h16* __restrict__ tf, float* __restrict__ causal, int P, int KP) {
    int p = blockIdx.x * blockDim.x + threadIdx.x;
    if (p >= P) return;
    float xv = x[p], tv = t[p];
    float pt = tv * tv - xv * xv;
    float prop = (pt > 0.f) ? sqrtf(pt) : 0.f;
    float cz = (fabsf(xv) <= tv) ? 1.f : 0.f;
    h16* row = tf + (size_t)p * KP;
    row[0] = (h16)xv; row[1] = (h16)tv; row[2] = (h16)prop; row[3] = (h16)cz;
    for (int c = 4; c < KP; ++c) row[c] = (h16)0.f;
    causal[p] = cz;
}

// ---------------- generic WMMA GEMM ----------------
// C(M,N) = act( A(MxK,f16,row-major) @ W(NxK,f16,row-major)^T * scale + bias ) * colMask
// One wave computes one 16x16 tile via v_wmma_f32_16x16x32_f16, f32 accumulate.
// K must be a multiple of 32; M,N multiples of 16.
__global__ void k_gemm_wmma(const h16* __restrict__ A, const h16* __restrict__ W,
                            const float* __restrict__ bias,
                            h16* __restrict__ outH, float* __restrict__ outF,
                            const float* __restrict__ colMask,
                            int M, int N, int K, float scale, int relu) {
    const int lane  = threadIdx.x & 31;
    const int wib   = threadIdx.x >> 5;
    const int wpb   = blockDim.x >> 5;
    const int ntn   = N >> 4;
    const int ntile = (M >> 4) * ntn;
    int tile = blockIdx.x * wpb + wib;
    if (tile >= ntile) return;            // wave-uniform: EXEC stays all-1s for WMMA

    const int mt = tile / ntn, nt = tile % ntn;
    const int m0 = mt << 4,  n0 = nt << 4;
    const int half = lane >> 4;           // which 16-lane half
    const int l15  = lane & 15;

    const h16* __restrict__ Arow = A + (size_t)(m0 + l15) * K;
    const h16* __restrict__ Wrow = W + (size_t)(n0 + l15) * K;
    __builtin_prefetch(Wrow, 0, 3);

    v8f acc = {};
    #pragma unroll 4
    for (int k0 = 0; k0 < K; k0 += 32) {
        v16h a, b;
        #pragma unroll
        for (int v = 0; v < 8; ++v) {
            // A 16x32 f16 fragment: K = (v/4)*16 + half*8 + (v%4)*2  (+0/+1)
            int ka = ((v >> 2) << 4) + (half << 3) + ((v & 3) << 1);
            // B 32x16 f16 fragment: K = (v/4)*16 + half*8 + (v%4)*2  (+0/+1), N = l15
            int kb = ((v >> 2) << 4) + (half << 3) + ((v & 3) << 1);
            a[2 * v]     = Arow[k0 + ka];
            a[2 * v + 1] = Arow[k0 + ka + 1];
            b[2 * v]     = Wrow[k0 + kb];
            b[2 * v + 1] = Wrow[k0 + kb + 1];
        }
        acc = __builtin_amdgcn_wmma_f32_16x16x32_f16(
            /*neg_a=*/false, a, /*neg_b=*/false, b,
            /*c_mod=*/(short)0, acc, /*reuse_a=*/false, /*reuse_b=*/false);
    }

    // C layout: VGPR r -> row = half*8 + r, col = l15
    const int col = n0 + l15;
    const float bb = bias    ? bias[col]    : 0.f;
    const float mk = colMask ? colMask[col] : 1.f;
    const int rowbase = m0 + (half << 3);
    #pragma unroll
    for (int r = 0; r < 8; ++r) {
        float val = acc[r] * scale + bb;
        if (relu) val = fmaxf(val, 0.f);
        val *= mk;
        size_t idx = (size_t)(rowbase + r) * N + col;
        if (outH) outH[idx] = (h16)val;
        if (outF) outF[idx] = val;
    }
}

// ---------------- launcher ----------------

static inline int ceil_div(int a, int b) { return (a + b - 1) / b; }

extern "C" void kernel_launch(void* const* d_in, const int* in_sizes, int n_in,
                              void* d_out, int out_size, void* d_ws, size_t ws_size,
                              hipStream_t stream) {
    const int B = 256, S = 1024, P = 4096, NM = 64, H = 128, PD = 128;
    const int K1 = 32; // trunk layer-1 K padded 4 -> 32

    const float* u0  = (const float*)d_in[0];
    const float* x   = (const float*)d_in[1];
    const float* t   = (const float*)d_in[2];
    const float* bw1 = (const float*)d_in[3];
    const float* bb1 = (const float*)d_in[4];
    const float* bw2 = (const float*)d_in[5];
    const float* bb2 = (const float*)d_in[6];
    const float* bw3 = (const float*)d_in[7];
    const float* bb3 = (const float*)d_in[8];
    const float* tw1 = (const float*)d_in[9];
    const float* tb1 = (const float*)d_in[10];
    const float* tw2 = (const float*)d_in[11];
    const float* tb2 = (const float*)d_in[12];
    const float* tw3 = (const float*)d_in[13];
    const float* tb3 = (const float*)d_in[14];
    float* out = (float*)d_out;

    // workspace carve-up (256B aligned)
    char* w = (char*)d_ws;
    auto alloc = [&](size_t bytes) -> void* {
        void* p = (void*)w;
        w += (bytes + 255) & ~(size_t)255;
        return p;
    };
    h16*   u0h   = (h16*)  alloc((size_t)B * S * 2);
    h16*   Wf    = (h16*)  alloc((size_t)2 * NM * S * 2);
    h16*   four  = (h16*)  alloc((size_t)B * 2 * NM * 2);
    h16*   bw1h  = (h16*)  alloc((size_t)H * 2 * NM * 2);
    h16*   bw2h  = (h16*)  alloc((size_t)H * H * 2);
    h16*   bw3h  = (h16*)  alloc((size_t)PD * H * 2);
    h16*   tw1h  = (h16*)  alloc((size_t)H * K1 * 2);
    h16*   tw2h  = (h16*)  alloc((size_t)H * H * 2);
    h16*   tw3h  = (h16*)  alloc((size_t)PD * H * 2);
    h16*   tfeat = (h16*)  alloc((size_t)P * K1 * 2);
    float* causal= (float*)alloc((size_t)P * 4);
    h16*   h1    = (h16*)  alloc((size_t)B * H * 2);
    h16*   h2    = (h16*)  alloc((size_t)B * H * 2);
    h16*   bro   = (h16*)  alloc((size_t)B * PD * 2);
    h16*   t1    = (h16*)  alloc((size_t)P * H * 2);
    h16*   t2    = (h16*)  alloc((size_t)P * H * 2);
    h16*   trk   = (h16*)  alloc((size_t)P * PD * 2);

    const int TPB = 256;

    // ---- prep ----
    k_f32_to_f16<<<ceil_div(B * S, TPB), TPB, 0, stream>>>(u0, u0h, B * S);
    k_build_trig<<<ceil_div(2 * NM * S, TPB), TPB, 0, stream>>>(Wf, S, NM);
    k_f32_to_f16<<<ceil_div(H * 2 * NM, TPB), TPB, 0, stream>>>(bw1, bw1h, H * 2 * NM);
    k_f32_to_f16<<<ceil_div(H * H, TPB), TPB, 0, stream>>>(bw2, bw2h, H * H);
    k_f32_to_f16<<<ceil_div(PD * H, TPB), TPB, 0, stream>>>(bw3, bw3h, PD * H);
    k_pad_convert<<<ceil_div(H * K1, TPB), TPB, 0, stream>>>(tw1, tw1h, H, 4, K1);
    k_f32_to_f16<<<ceil_div(H * H, TPB), TPB, 0, stream>>>(tw2, tw2h, H * H);
    k_f32_to_f16<<<ceil_div(PD * H, TPB), TPB, 0, stream>>>(tw3, tw3h, PD * H);
    k_build_tfeat<<<ceil_div(P, TPB), TPB, 0, stream>>>(x, t, tfeat, causal, P, K1);

    auto gemm = [&](const h16* A, const h16* W_, const float* bias,
                    h16* oh, float* of, const float* mask,
                    int M, int N, int K, float scale, int relu) {
        int ntiles = (M >> 4) * (N >> 4);
        int blocks = ceil_div(ntiles, TPB / 32);
        k_gemm_wmma<<<blocks, TPB, 0, stream>>>(A, W_, bias, oh, of, mask,
                                                M, N, K, scale, relu);
    };

    // ---- Fourier coefficients: (B x S) @ trig^T / S ----
    gemm(u0h, Wf, nullptr, four, nullptr, nullptr, B, 2 * NM, S, 1.0f / (float)S, 0);

    // ---- branch MLP (256 rows) ----
    gemm(four, bw1h, bb1, h1, nullptr, nullptr, B, H, 2 * NM, 1.f, 1);
    gemm(h1,   bw2h, bb2, h2, nullptr, nullptr, B, H, H,      1.f, 1);
    gemm(h2,   bw3h, bb3, bro, nullptr, nullptr, B, PD, H,    1.f, 0);

    // ---- trunk MLP (4096 rows, batch-independent) ----
    gemm(tfeat, tw1h, tb1, t1, nullptr, nullptr, P, H, K1, 1.f, 1);
    gemm(t1,    tw2h, tb2, t2, nullptr, nullptr, P, H, H,  1.f, 1);
    gemm(t2,    tw3h, tb3, trk, nullptr, nullptr, P, PD, H, 1.f, 0);

    // ---- final einsum + causal mask: (B x PD) @ trunk^T, f32 out ----
    gemm(bro, trk, nullptr, nullptr, out, causal, B, P, PD, 1.f, 0);
}
